// BipartiteGCN_Cl_61074434949684
// MI455X (gfx1250) — matrix operand order repaired and checked by reading, not compiled, not run on hardware
//
#include <hip/hip_runtime.h>
#include <cstdint>
#include <cstddef>

#define EMB 64

typedef __attribute__((ext_vector_type(16))) _Float16 v16h;
typedef __attribute__((ext_vector_type(8)))  _Float16 v8h;
typedef __attribute__((ext_vector_type(8)))  float    v8f;

// ---------------- d_in index map (pytree sorted-key order) ----------------
// 0 cons_feats [Nc,5]  1 edge_index [2,E] (row0=cons,row1=var)  2 edge_feats [E,1]  3 var_feats [Nv,19]
enum {
  P_cW1 = 4, P_cW2, P_cb1, P_cb2, P_cln_b, P_cln_w,
  CV_fe_W = 10, CV_ff_W, CV_ff_b, CV_ff_ln_b, CV_ff_ln_w, CV_fl_W, CV_fl_b, CV_fr_W,
  CV_out_W1, CV_out_W2, CV_out_b1, CV_out_b2, CV_pc_ln_b, CV_pc_ln_w,
  P_eln_b = 24, P_eln_w, P_hW1, P_hW2, P_hb1, P_hb2,
  P_vW1 = 30, P_vW2, P_vb1, P_vb2,
  VC_fe_W = 34, VC_ff_W, VC_ff_b, VC_ff_ln_b, VC_ff_ln_w, VC_fl_W, VC_fl_b, VC_fr_W,
  VC_out_W1, VC_out_W2, VC_out_b1, VC_out_b2, VC_pc_ln_b, VC_pc_ln_w,
  P_vln_b = 48, P_vln_w = 49
};

__device__ __forceinline__ float leaky01(float x) { return x >= 0.f ? x : 0.01f * x; }

// Wave-internal LDS producer->consumer ordering (same wave writes then reads LDS).
__device__ __forceinline__ void wave_lds_sync() {
  __builtin_amdgcn_wave_barrier();
  asm volatile("s_wait_dscnt 0" ::: "memory");
  __builtin_amdgcn_wave_barrier();
}

// A-fragment (16x32 f16 slice of a 16 x rowStride LDS tile), CDNA5 layout:
// lanes 0-15: row=lane, K = k0+0..7 and k0+16..23 ; lanes 16-31: row=lane-16, K = k0+8..15 and k0+24..31
__device__ __forceinline__ v16h load_a_frag(const _Float16* sA, int lane, int k0, int rowStride) {
  int r = lane & 15;
  const _Float16* p = sA + r * rowStride + k0 + ((lane & 16) ? 8 : 0);
  v8h lo = *(const v8h*)p;
  v8h hi = *(const v8h*)(p + 16);
  return __builtin_shufflevector(lo, hi, 0,1,2,3,4,5,6,7,8,9,10,11,12,13,14,15);
}

// B-fragment from transposed weights Wt[n][k] in LDS:
// lanes 0-15: N=n0+lane, K=k0+0..7 and k0+16..23 ; lanes 16-31: N=n0+lane-16, K=k0+8..15 and k0+24..31
__device__ __forceinline__ v16h load_b_frag(const _Float16* sWt, int lane, int k0, int n0, int kStride) {
  int n = n0 + (lane & 15);
  int kb = k0 + ((lane & 16) ? 8 : 0);
  const _Float16* p = sWt + n * kStride + kb;
  v8h lo = *(const v8h*)p;
  v8h hi = *(const v8h*)(p + 16);
  return __builtin_shufflevector(lo, hi, 0,1,2,3,4,5,6,7,8,9,10,11,12,13,14,15);
}

// ---------------------------------------------------------------------------
__global__ void zero_f32_kernel(float* __restrict__ p, long long n) {
  long long i = (long long)blockIdx.x * blockDim.x + threadIdx.x;
  long long s = (long long)gridDim.x * blockDim.x;
  for (; i < n; i += s) p[i] = 0.f;
}

// ---------------------------------------------------------------------------
// Node embedding: LN(FIN) -> Linear(FIN,64)+leaky (scalar, tiny K) -> Linear(64,64)+leaky (WMMA).
// 16 rows per wave, 8 waves/block.
template <int FIN>
__global__ void embed_nodes_wmma_kernel(const float* __restrict__ x,
                                        const float* __restrict__ lnw, const float* __restrict__ lnb,
                                        const float* __restrict__ W1, const float* __restrict__ b1,
                                        const float* __restrict__ W2, const float* __restrict__ b2,
                                        float* __restrict__ out, int n) {
  __shared__ float sW1[FIN * EMB];
  __shared__ __attribute__((aligned(32))) _Float16 sW2t[EMB * EMB];  // W2t[n][k]
  __shared__ __attribute__((aligned(32))) _Float16 sA[8][16 * EMB];
  __shared__ float sb1[EMB], sb2[EMB], sLw[FIN], sLb[FIN];
  int tid = threadIdx.x;
  for (int i = tid; i < FIN * EMB; i += blockDim.x) sW1[i] = W1[i];
  for (int i = tid; i < EMB * EMB; i += blockDim.x) {
    int k = i >> 6, nn = i & 63;
    sW2t[nn * EMB + k] = (_Float16)W2[i];
  }
  if (tid < EMB) { sb1[tid] = b1[tid]; sb2[tid] = b2[tid]; }
  if (tid < FIN) { sLw[tid] = lnw[tid]; sLb[tid] = lnb[tid]; }
  __syncthreads();

  int w = tid >> 5, lane = tid & 31;
  int row0 = (blockIdx.x * 8 + w) * 16;
  _Float16* a = sA[w];
  {  // phase 1: LN + Linear(FIN,64) + leaky -> f16 tile. Lane pair: row=lane>>1, cols=(lane&1)*32..+32.
    int r = lane >> 1, c0 = (lane & 1) * 32;
    int row = row0 + r;
    _Float16* d = a + r * EMB + c0;
    if (row < n) {
      float xi[FIN];
      float mu = 0.f;
#pragma unroll
      for (int i = 0; i < FIN; ++i) { xi[i] = x[(size_t)row * FIN + i]; mu += xi[i]; }
      mu *= (1.f / FIN);
      float var = 0.f;
#pragma unroll
      for (int i = 0; i < FIN; ++i) { float dd = xi[i] - mu; var += dd * dd; }
      var *= (1.f / FIN);
      float rstd = rsqrtf(var + 1e-5f);
#pragma unroll
      for (int i = 0; i < FIN; ++i) xi[i] = (xi[i] - mu) * rstd * sLw[i] + sLb[i];
#pragma unroll 4
      for (int j = 0; j < 32; ++j) {
        float acc1 = sb1[c0 + j];
#pragma unroll
        for (int i = 0; i < FIN; ++i) acc1 += xi[i] * sW1[i * EMB + c0 + j];
        d[j] = (_Float16)leaky01(acc1);
      }
    } else {
#pragma unroll
      for (int j = 0; j < 32; ++j) d[j] = (_Float16)0.f;
    }
  }
  wave_lds_sync();
  int lc = lane & 15;
  int mb = (lane & 16) ? 8 : 0;
  bool fullTile = (row0 + 16 <= n);
  v16h a0 = load_a_frag(a, lane, 0, EMB);
  v16h a1 = load_a_frag(a, lane, 32, EMB);
#pragma unroll
  for (int n0 = 0; n0 < 4; ++n0) {
    float bj = sb2[n0 * 16 + lc];
    v8f acc;
#pragma unroll
    for (int i = 0; i < 8; ++i) acc[i] = bj;
    v16h b0 = load_b_frag(sW2t, lane, 0, n0 * 16, EMB);
    v16h b1f = load_b_frag(sW2t, lane, 32, n0 * 16, EMB);
    acc = __builtin_amdgcn_wmma_f32_16x16x32_f16(false, a0, false, b0, (short)0, acc, false, false);
    acc = __builtin_amdgcn_wmma_f32_16x16x32_f16(false, a1, false, b1f, (short)0, acc, false, false);
    float* po = out + (size_t)(row0 + mb) * EMB + n0 * 16 + lc;
    if (fullTile) {
#pragma unroll
      for (int i = 0; i < 8; ++i) po[i * EMB] = leaky01(acc[i]);
    } else {
#pragma unroll
      for (int i = 0; i < 8; ++i)
        if (row0 + mb + i < n) po[i * EMB] = leaky01(acc[i]);
    }
  }
}

// ---------------------------------------------------------------------------
// out[n,64] = in[n,64] @ W[64,64] (+ bias); WMMA, 16 rows per wave, 8 waves/block.
__global__ void gemm64_wmma_kernel(const float* __restrict__ in, const float* __restrict__ W,
                                   const float* __restrict__ bias, float* __restrict__ out,
                                   int n) {
  __shared__ __attribute__((aligned(32))) _Float16 sWt[EMB * EMB];   // Wt[n][k]
  __shared__ __attribute__((aligned(32))) _Float16 sA[8][16 * EMB];
  __shared__ float sB[EMB];
  int tid = threadIdx.x;
  for (int i = tid; i < EMB * EMB; i += blockDim.x) {
    int k = i >> 6, nn = i & 63;
    sWt[nn * EMB + k] = (_Float16)W[i];
  }
  if (tid < EMB) sB[tid] = bias ? bias[tid] : 0.f;
  __syncthreads();

  int w = tid >> 5, lane = tid & 31;
  int row0 = (blockIdx.x * 8 + w) * 16;
  _Float16* a = sA[w];
  {  // lane pair (2 lanes per row) converts f32 rows -> f16 LDS tile
    int r = lane >> 1, c0 = (lane & 1) * 32;
    int row = row0 + r;
    _Float16* d = a + r * EMB + c0;
    if (row < n) {
      const float4* p = (const float4*)(in + (size_t)row * EMB + c0);
#pragma unroll
      for (int q = 0; q < 8; ++q) {
        float4 v = p[q];
        d[q * 4 + 0] = (_Float16)v.x; d[q * 4 + 1] = (_Float16)v.y;
        d[q * 4 + 2] = (_Float16)v.z; d[q * 4 + 3] = (_Float16)v.w;
      }
    } else {
#pragma unroll
      for (int q = 0; q < 32; ++q) d[q] = (_Float16)0.f;
    }
  }
  wave_lds_sync();
  int lc = lane & 15;
  int mb = (lane & 16) ? 8 : 0;
  bool fullTile = (row0 + 16 <= n);
  v16h a0 = load_a_frag(a, lane, 0, EMB);
  v16h a1 = load_a_frag(a, lane, 32, EMB);
#pragma unroll
  for (int n0 = 0; n0 < 4; ++n0) {
    float bj = sB[n0 * 16 + lc];
    v8f acc;
#pragma unroll
    for (int i = 0; i < 8; ++i) acc[i] = bj;
    v16h b0 = load_b_frag(sWt, lane, 0, n0 * 16, EMB);
    v16h b1 = load_b_frag(sWt, lane, 32, n0 * 16, EMB);
    acc = __builtin_amdgcn_wmma_f32_16x16x32_f16(false, a0, false, b0, (short)0, acc, false, false);
    acc = __builtin_amdgcn_wmma_f32_16x16x32_f16(false, a1, false, b1, (short)0, acc, false, false);
    float* po = out + (size_t)(row0 + mb) * EMB + n0 * 16 + lc;
    if (fullTile) {
#pragma unroll
      for (int i = 0; i < 8; ++i) po[i * EMB] = acc[i];
    } else {
#pragma unroll
      for (int i = 0; i < 8; ++i)
        if (row0 + mb + i < n) po[i * EMB] = acc[i];
    }
  }
}

// ---------------------------------------------------------------------------
// Fused edge phase: z = L[src] + R[dst] + cvec ; LN ; relu ; z @ ffW + ffb ; atomic scatter to agg[dst].
// 16 edges per wave, 8 waves/block (128 edges/block). cvec = eln_b[0]*fe_W (LN of width-1 == bias).
__global__ void edge_conv_kernel(const float* __restrict__ L, const float* __restrict__ R,
                                 const int* __restrict__ src, const int* __restrict__ dst,
                                 const float* __restrict__ eln_b, const float* __restrict__ fe_W,
                                 const float* __restrict__ lnw, const float* __restrict__ lnb,
                                 const float* __restrict__ ffW, const float* __restrict__ ffb,
                                 float* __restrict__ agg, int nE) {
  __shared__ __attribute__((aligned(32))) _Float16 sWt[EMB * EMB];
  __shared__ __attribute__((aligned(32))) _Float16 sA[8][16 * EMB];
  __shared__ float sB[EMB], sCvec[EMB], sLw[EMB], sLb[EMB];
  __shared__ __attribute__((aligned(16))) int sSrc[8][16], sDst[8][16];
  int tid = threadIdx.x;
  for (int i = tid; i < EMB * EMB; i += blockDim.x) {
    int k = i >> 6, nn = i & 63;
    sWt[nn * EMB + k] = (_Float16)ffW[i];
  }
  if (tid < EMB) {
    sB[tid] = ffb[tid];
    sCvec[tid] = eln_b[0] * fe_W[tid];
    sLw[tid] = lnw[tid]; sLb[tid] = lnb[tid];
  }
  __syncthreads();

  int w = tid >> 5, lane = tid & 31;
  int e0 = (blockIdx.x * 8 + w) * 16;
  if (lane < 16) {
    int e = e0 + lane;
    sSrc[w][lane] = (e < nE) ? src[e] : -1;
    sDst[w][lane] = (e < nE) ? dst[e] : -1;
  }
  wave_lds_sync();
  _Float16* a = sA[w];
  {  // gather + LN + relu -> f16 tile; lane pair per edge row
    int r = lane >> 1, c0 = (lane & 1) * 32;
    int s = sSrc[w][r], d = sDst[w][r];
    float z[32];
    if (s >= 0) {
      const float4* lp = (const float4*)(L + (size_t)s * EMB + c0);
      const float4* rp = (const float4*)(R + (size_t)d * EMB + c0);
#pragma unroll
      for (int q = 0; q < 8; ++q) {
        float4 lv = lp[q], rv = rp[q];
        z[q * 4 + 0] = lv.x + rv.x + sCvec[c0 + q * 4 + 0];
        z[q * 4 + 1] = lv.y + rv.y + sCvec[c0 + q * 4 + 1];
        z[q * 4 + 2] = lv.z + rv.z + sCvec[c0 + q * 4 + 2];
        z[q * 4 + 3] = lv.w + rv.w + sCvec[c0 + q * 4 + 3];
      }
    } else {
#pragma unroll
      for (int q = 0; q < 32; ++q) z[q] = 0.f;
    }
    float s1 = 0.f, s2 = 0.f;
#pragma unroll
    for (int q = 0; q < 32; ++q) { s1 += z[q]; s2 += z[q] * z[q]; }
    s1 += __shfl_xor(s1, 1, 32);
    s2 += __shfl_xor(s2, 1, 32);
    float mu = s1 * (1.f / EMB);
    float var = s2 * (1.f / EMB) - mu * mu;
    float rstd = rsqrtf(var + 1e-5f);
    _Float16* drow = a + r * EMB + c0;
#pragma unroll
    for (int q = 0; q < 32; ++q) {
      float hv = (z[q] - mu) * rstd * sLw[c0 + q] + sLb[c0 + q];
      drow[q] = (_Float16)fmaxf(hv, 0.f);
    }
  }
  wave_lds_sync();
  int lc = lane & 15;
  int mb = (lane & 16) ? 8 : 0;
  // Hoist scatter targets out of the n-tile loop: dst rows are n0-invariant.
  // Branch-free tail: invalid edges scatter 0.0f into row 0 (numeric no-op).
  float* pb[8];
  float msk[8];
#pragma unroll
  for (int i = 0; i < 8; ++i) {
    int dd = sDst[w][mb + i];
    msk[i] = (dd < 0) ? 0.f : 1.f;
    pb[i] = agg + (size_t)(dd < 0 ? 0 : dd) * EMB + lc;
  }
  v16h a0 = load_a_frag(a, lane, 0, EMB);
  v16h a1 = load_a_frag(a, lane, 32, EMB);
#pragma unroll
  for (int n0 = 0; n0 < 4; ++n0) {
    float bj = sB[n0 * 16 + lc];
    v8f acc;
#pragma unroll
    for (int i = 0; i < 8; ++i) acc[i] = bj;
    v16h b0 = load_b_frag(sWt, lane, 0, n0 * 16, EMB);
    v16h b1 = load_b_frag(sWt, lane, 32, n0 * 16, EMB);
    acc = __builtin_amdgcn_wmma_f32_16x16x32_f16(false, a0, false, b0, (short)0, acc, false, false);
    acc = __builtin_amdgcn_wmma_f32_16x16x32_f16(false, a1, false, b1, (short)0, acc, false, false);
#pragma unroll
    for (int i = 0; i < 8; ++i)
      atomicAdd(pb[i] + n0 * 16, acc[i] * msk[i]);
  }
}

// ---------------------------------------------------------------------------
// Post-conv: h = concat(LN(agg), right) [16x128]; relu(h@W1+b1) [16x64]; @W2+b2 -> out.
// 16 rows per wave, 4 waves/block.
__global__ void post_conv_kernel(const float* __restrict__ agg, const float* __restrict__ right,
                                 const float* __restrict__ lnw, const float* __restrict__ lnb,
                                 const float* __restrict__ W1, const float* __restrict__ b1,
                                 const float* __restrict__ W2, const float* __restrict__ b2,
                                 float* __restrict__ out, int n) {
  __shared__ __attribute__((aligned(32))) _Float16 sW1t[EMB * 128];  // Wt1[n][k], k in 0..127
  __shared__ __attribute__((aligned(32))) _Float16 sW2t[EMB * EMB];
  __shared__ __attribute__((aligned(32))) _Float16 sH[4][16 * 128];
  __shared__ __attribute__((aligned(32))) _Float16 sH2[4][16 * EMB];
  __shared__ float sB1[EMB], sB2[EMB], sLw[EMB], sLb[EMB];
  int tid = threadIdx.x;
  for (int i = tid; i < 128 * EMB; i += blockDim.x) {
    int k = i >> 6, nn = i & 63;
    sW1t[nn * 128 + k] = (_Float16)W1[i];
  }
  for (int i = tid; i < EMB * EMB; i += blockDim.x) {
    int k = i >> 6, nn = i & 63;
    sW2t[nn * EMB + k] = (_Float16)W2[i];
  }
  if (tid < EMB) { sB1[tid] = b1[tid]; sB2[tid] = b2[tid]; sLw[tid] = lnw[tid]; sLb[tid] = lnb[tid]; }
  __syncthreads();

  int w = tid >> 5, lane = tid & 31;
  int row0 = (blockIdx.x * 4 + w) * 16;
  _Float16* h = sH[w];
  {  // LN(agg) -> cols 0..63 ; right -> cols 64..127
    int r = lane >> 1, c0 = (lane & 1) * 32;
    int row = row0 + r;
    float z[32];
    if (row < n) {
      const float4* p = (const float4*)(agg + (size_t)row * EMB + c0);
#pragma unroll
      for (int q = 0; q < 8; ++q) {
        float4 v = p[q];
        z[q * 4 + 0] = v.x; z[q * 4 + 1] = v.y; z[q * 4 + 2] = v.z; z[q * 4 + 3] = v.w;
      }
    } else {
#pragma unroll
      for (int q = 0; q < 32; ++q) z[q] = 0.f;
    }
    float s1 = 0.f, s2 = 0.f;
#pragma unroll
    for (int q = 0; q < 32; ++q) { s1 += z[q]; s2 += z[q] * z[q]; }
    s1 += __shfl_xor(s1, 1, 32);
    s2 += __shfl_xor(s2, 1, 32);
    float mu = s1 * (1.f / EMB);
    float var = s2 * (1.f / EMB) - mu * mu;
    float rstd = rsqrtf(var + 1e-5f);
    _Float16* drow = h + r * 128 + c0;
#pragma unroll
    for (int q = 0; q < 32; ++q)
      drow[q] = (_Float16)((z[q] - mu) * rstd * sLw[c0 + q] + sLb[c0 + q]);
    _Float16* drow2 = h + r * 128 + 64 + c0;
    if (row < n) {
      const float4* rp = (const float4*)(right + (size_t)row * EMB + c0);
#pragma unroll
      for (int q = 0; q < 8; ++q) {
        float4 v = rp[q];
        drow2[q * 4 + 0] = (_Float16)v.x; drow2[q * 4 + 1] = (_Float16)v.y;
        drow2[q * 4 + 2] = (_Float16)v.z; drow2[q * 4 + 3] = (_Float16)v.w;
      }
    } else {
#pragma unroll
      for (int q = 0; q < 32; ++q) drow2[q] = (_Float16)0.f;
    }
  }
  wave_lds_sync();
  int lc = lane & 15;
  int mb = (lane & 16) ? 8 : 0;
  bool fullTile = (row0 + 16 <= n);
  // GEMM1: [16x128] @ [128x64] + b1, relu -> sH2 (f16)
  _Float16* h2 = sH2[w];
  v16h af[4];
#pragma unroll
  for (int kk = 0; kk < 4; ++kk) af[kk] = load_a_frag(h, lane, kk * 32, 128);
#pragma unroll
  for (int n0 = 0; n0 < 4; ++n0) {
    float bj = sB1[n0 * 16 + lc];
    v8f acc;
#pragma unroll
    for (int i = 0; i < 8; ++i) acc[i] = bj;
#pragma unroll
    for (int kk = 0; kk < 4; ++kk) {
      v16h bf = load_b_frag(sW1t, lane, kk * 32, n0 * 16, 128);
      acc = __builtin_amdgcn_wmma_f32_16x16x32_f16(false, af[kk], false, bf, (short)0, acc, false, false);
    }
    _Float16* ph = h2 + mb * EMB + n0 * 16 + lc;
#pragma unroll
    for (int i = 0; i < 8; ++i)
      ph[i * EMB] = (_Float16)fmaxf(acc[i], 0.f);
  }
  wave_lds_sync();
  // GEMM2: [16x64] @ [64x64] + b2 -> out
  v16h c0f = load_a_frag(h2, lane, 0, EMB);
  v16h c1f = load_a_frag(h2, lane, 32, EMB);
#pragma unroll
  for (int n0 = 0; n0 < 4; ++n0) {
    float bj = sB2[n0 * 16 + lc];
    v8f acc;
#pragma unroll
    for (int i = 0; i < 8; ++i) acc[i] = bj;
    v16h b0 = load_b_frag(sW2t, lane, 0, n0 * 16, EMB);
    v16h b1f = load_b_frag(sW2t, lane, 32, n0 * 16, EMB);
    acc = __builtin_amdgcn_wmma_f32_16x16x32_f16(false, c0f, false, b0, (short)0, acc, false, false);
    acc = __builtin_amdgcn_wmma_f32_16x16x32_f16(false, c1f, false, b1f, (short)0, acc, false, false);
    float* po = out + (size_t)(row0 + mb) * EMB + n0 * 16 + lc;
    if (fullTile) {
#pragma unroll
      for (int i = 0; i < 8; ++i) po[i * EMB] = acc[i];
    } else {
#pragma unroll
      for (int i = 0; i < 8; ++i)
        if (row0 + mb + i < n) po[i * EMB] = acc[i];
    }
  }
}

// ---------------------------------------------------------------------------
// Head (WMMA): h = leaky(v@hW1+hb1) via 8 WMMAs; out[row] = hb2 + sum_col h[row][col]*hW2[col]
// folded into the epilogue with a 4-step shfl_xor butterfly over the 16 columns per half-wave.
__global__ void head_wmma_kernel(const float* __restrict__ v,
                                 const float* __restrict__ W1, const float* __restrict__ b1,
                                 const float* __restrict__ W2, const float* __restrict__ b2,
                                 float* __restrict__ out, int n) {
  __shared__ __attribute__((aligned(32))) _Float16 sW1t[EMB * EMB];  // W1t[n][k]
  __shared__ __attribute__((aligned(32))) _Float16 sA[8][16 * EMB];
  __shared__ float sB1[EMB], sW2[EMB];
  int tid = threadIdx.x;
  for (int i = tid; i < EMB * EMB; i += blockDim.x) {
    int k = i >> 6, nn = i & 63;
    sW1t[nn * EMB + k] = (_Float16)W1[i];
  }
  if (tid < EMB) { sB1[tid] = b1[tid]; sW2[tid] = W2[tid]; }  // W2 is [64,1]
  __syncthreads();

  int w = tid >> 5, lane = tid & 31;
  int row0 = (blockIdx.x * 8 + w) * 16;
  _Float16* a = sA[w];
  {  // gather v rows f32 -> f16 tile (lane pair)
    int r = lane >> 1, c0 = (lane & 1) * 32;
    int row = row0 + r;
    _Float16* d = a + r * EMB + c0;
    if (row < n) {
      const float4* p = (const float4*)(v + (size_t)row * EMB + c0);
#pragma unroll
      for (int q = 0; q < 8; ++q) {
        float4 t = p[q];
        d[q * 4 + 0] = (_Float16)t.x; d[q * 4 + 1] = (_Float16)t.y;
        d[q * 4 + 2] = (_Float16)t.z; d[q * 4 + 3] = (_Float16)t.w;
      }
    } else {
#pragma unroll
      for (int q = 0; q < 32; ++q) d[q] = (_Float16)0.f;
    }
  }
  wave_lds_sync();
  int lc = lane & 15;
  int mb = (lane & 16) ? 8 : 0;
  v16h a0 = load_a_frag(a, lane, 0, EMB);
  v16h a1 = load_a_frag(a, lane, 32, EMB);
  float partial[8];
#pragma unroll
  for (int i = 0; i < 8; ++i) partial[i] = 0.f;
#pragma unroll
  for (int n0 = 0; n0 < 4; ++n0) {
    float bj = sB1[n0 * 16 + lc];
    v8f acc;
#pragma unroll
    for (int i = 0; i < 8; ++i) acc[i] = bj;
    v16h b0 = load_b_frag(sW1t, lane, 0, n0 * 16, EMB);
    v16h b1f = load_b_frag(sW1t, lane, 32, n0 * 16, EMB);
    acc = __builtin_amdgcn_wmma_f32_16x16x32_f16(false, a0, false, b0, (short)0, acc, false, false);
    acc = __builtin_amdgcn_wmma_f32_16x16x32_f16(false, a1, false, b1f, (short)0, acc, false, false);
    float w2 = sW2[n0 * 16 + lc];
#pragma unroll
    for (int i = 0; i < 8; ++i) partial[i] += leaky01(acc[i]) * w2;
  }
  // reduce over the 16 columns held by each half-wave (xor masks stay within the half)
#pragma unroll
  for (int m = 1; m < 16; m <<= 1) {
#pragma unroll
    for (int i = 0; i < 8; ++i) partial[i] += __shfl_xor(partial[i], m, 32);
  }
  float b2v = b2[0];
  if (lc == 0) {  // lanes 0 and 16 hold rows row0+mb .. row0+mb+7
    float* po = out + row0 + mb;
    if (row0 + 16 <= n) {
      float4 o0 = make_float4(partial[0] + b2v, partial[1] + b2v, partial[2] + b2v, partial[3] + b2v);
      float4 o1 = make_float4(partial[4] + b2v, partial[5] + b2v, partial[6] + b2v, partial[7] + b2v);
      *(float4*)po = o0;
      *(float4*)(po + 4) = o1;
    } else {
#pragma unroll
      for (int i = 0; i < 8; ++i)
        if (row0 + mb + i < n) po[i] = partial[i] + b2v;
    }
  }
}

// ---------------------------------------------------------------------------
extern "C" void kernel_launch(void* const* d_in, const int* in_sizes, int n_in,
                              void* d_out, int out_size, void* d_ws, size_t ws_size,
                              hipStream_t stream) {
  const float* cons  = (const float*)d_in[0];
  const int*   eidx  = (const int*)d_in[1];
  const float* vfeat = (const float*)d_in[3];
  int Nc = in_sizes[0] / 5;
  int E  = in_sizes[2];
  int Nv = in_sizes[3] / 19;
  const int* ei_c = eidx;      // row 0: cons indices
  const int* ei_v = eidx + E;  // row 1: var indices

  auto P = [&](int i) { return (const float*)d_in[i]; };

  // workspace carve (256B-aligned f32 buffers)
  char* ws = (char*)d_ws;
  size_t off = 0;
  auto carve = [&](size_t elems) {
    float* p = (float*)(ws + off);
    off += ((elems * sizeof(float) + 255) / 256) * 256;
    return p;
  };
  int Nmax = Nc > Nv ? Nc : Nv;
  float* c_emb = carve((size_t)Nc * EMB);
  float* v_emb = carve((size_t)Nv * EMB);
  float* Lbuf  = carve((size_t)Nmax * EMB);
  float* Rbuf  = carve((size_t)Nmax * EMB);
  float* agg   = carve((size_t)Nmax * EMB);
  float* c2    = carve((size_t)Nc * EMB);
  float* v2    = carve((size_t)Nv * EMB);
  (void)ws_size; (void)n_in; (void)out_size;

  dim3 b256(256), b128(128);
  int gE = (E + 127) / 128;

  // node embeddings (WMMA second layer)
  embed_nodes_wmma_kernel<5><<<(Nc + 127) / 128, b256, 0, stream>>>(
      cons, P(P_cln_w), P(P_cln_b), P(P_cW1), P(P_cb1), P(P_cW2), P(P_cb2), c_emb, Nc);
  embed_nodes_wmma_kernel<19><<<(Nv + 127) / 128, b256, 0, stream>>>(
      vfeat, P(P_vln_w), P(P_vln_b), P(P_vW1), P(P_vb1), P(P_vW2), P(P_vb2), v_emb, Nv);

  // ---- conv vc: left=vars, right=cons, dst=cons ----
  gemm64_wmma_kernel<<<(Nv + 127) / 128, b256, 0, stream>>>(v_emb, P(VC_fr_W), nullptr, Lbuf, Nv);
  gemm64_wmma_kernel<<<(Nc + 127) / 128, b256, 0, stream>>>(c_emb, P(VC_fl_W), P(VC_fl_b), Rbuf, Nc);
  zero_f32_kernel<<<2048, b256, 0, stream>>>(agg, (long long)Nc * EMB);
  edge_conv_kernel<<<gE, b256, 0, stream>>>(Lbuf, Rbuf, ei_v, ei_c,
      P(P_eln_b), P(VC_fe_W), P(VC_ff_ln_w), P(VC_ff_ln_b), P(VC_ff_W), P(VC_ff_b), agg, E);
  post_conv_kernel<<<(Nc + 63) / 64, b128, 0, stream>>>(agg, c_emb,
      P(VC_pc_ln_w), P(VC_pc_ln_b), P(VC_out_W1), P(VC_out_b1), P(VC_out_W2), P(VC_out_b2), c2, Nc);

  // ---- conv cv: left=cons(c2), right=vars, dst=vars ----
  gemm64_wmma_kernel<<<(Nc + 127) / 128, b256, 0, stream>>>(c2, P(CV_fr_W), nullptr, Lbuf, Nc);
  gemm64_wmma_kernel<<<(Nv + 127) / 128, b256, 0, stream>>>(v_emb, P(CV_fl_W), P(CV_fl_b), Rbuf, Nv);
  zero_f32_kernel<<<2048, b256, 0, stream>>>(agg, (long long)Nv * EMB);
  edge_conv_kernel<<<gE, b256, 0, stream>>>(Lbuf, Rbuf, ei_c, ei_v,
      P(P_eln_b), P(CV_fe_W), P(CV_ff_ln_w), P(CV_ff_ln_b), P(CV_ff_W), P(CV_ff_b), agg, E);
  post_conv_kernel<<<(Nv + 63) / 64, b128, 0, stream>>>(agg, v_emb,
      P(CV_pc_ln_w), P(CV_pc_ln_b), P(CV_out_W1), P(CV_out_b1), P(CV_out_W2), P(CV_out_b2), v2, Nv);

  // head (WMMA)
  head_wmma_kernel<<<(Nv + 127) / 128, b256, 0, stream>>>(
      v2, P(P_hW1), P(P_hb1), P(P_hW2), P(P_hb2), (float*)d_out, Nv);
}
